// vGINMolEncoder_86904368268093
// MI455X (gfx1250) — compile-verified, matrix operand order checked by reading, not run on hardware
//
#include <hip/hip_runtime.h>

#define H    300
#define HP   320
#define H2   600
#define H2P  640
#define NLAYER 5

typedef __attribute__((ext_vector_type(16))) _Float16 v16h;
typedef __attribute__((ext_vector_type(8)))  _Float16 v8h;
typedef __attribute__((ext_vector_type(8)))  float    v8f;

// ---------------------------------------------------------------------------
// Elementwise / gather / scatter kernels  (block = node/edge, 320 threads)
// ---------------------------------------------------------------------------
__global__ void k_gather(const int* __restrict__ idx, const float* __restrict__ table,
                         float* __restrict__ out) {
    int n = blockIdx.x, c = threadIdx.x;
    if (c >= H) return;
    out[(long long)n * H + c] = table[(long long)idx[n] * H + c];
}

__global__ void k_init_vn(const float* __restrict__ vemb, float* __restrict__ vn) {
    int b = blockIdx.x, c = threadIdx.x;
    if (c >= H) return;
    vn[(long long)b * H + c] = vemb[c];
}

__global__ void k_addvn(const float* __restrict__ h, const float* __restrict__ vn,
                        const int* __restrict__ batch,
                        float* __restrict__ hin, float* __restrict__ aggr) {
    int n = blockIdx.x, c = threadIdx.x;
    if (c >= H) return;
    int b = batch[n];
    long long o = (long long)n * H + c;
    hin[o]  = h[o] + vn[(long long)b * H + c];
    aggr[o] = 0.0f;
}

__global__ void k_message(const int* __restrict__ src, const int* __restrict__ dst,
                          const float* __restrict__ hin, const float* __restrict__ ef,
                          float* __restrict__ aggr) {
    int e = blockIdx.x, c = threadIdx.x;
    if (c >= H) return;
    int s = src[e], d = dst[e];
    float m = hin[(long long)s * H + c] + ef[(long long)e * H + c];
    m = fmaxf(m, 0.0f);
    unsafeAtomicAdd(&aggr[(long long)d * H + c], m);   // global_atomic_add_f32
}

__global__ void k_pre_cast(const float* __restrict__ hin, const float* __restrict__ aggr,
                           _Float16* __restrict__ out) {
    int n = blockIdx.x, k = threadIdx.x;                // 320 threads
    float v = 0.0f;
    if (k < H) {
        long long o = (long long)n * H + k;
        v = hin[o] + aggr[o];
    }
    out[(long long)n * HP + k] = (_Float16)v;
}

__global__ void k_pool(const int* __restrict__ batch, const float* __restrict__ h,
                       float* __restrict__ pooled) {
    int n = blockIdx.x, c = threadIdx.x;
    if (c >= H) return;
    unsafeAtomicAdd(&pooled[(long long)batch[n] * H + c], h[(long long)n * H + c]);
}

__global__ void k_pooled_cast(const float* __restrict__ pooled, const float* __restrict__ vn,
                              _Float16* __restrict__ out) {
    int b = blockIdx.x, k = threadIdx.x;                // 320 threads
    float v = 0.0f;
    if (k < H) v = pooled[(long long)b * H + k] + vn[(long long)b * H + k];
    out[(long long)b * HP + k] = (_Float16)v;
}

__global__ void k_castpad(const float* __restrict__ X, int rin, int cin, int ldin,
                          _Float16* __restrict__ Y, int cout) {
    int r = blockIdx.x, c = threadIdx.x;
    if (c >= cout) return;
    float v = (r < rin && c < cin) ? X[(long long)r * ldin + c] : 0.0f;
    Y[(long long)r * cout + c] = (_Float16)v;
}

__global__ void k_zero(float* __restrict__ p, long long n) {
    long long i = (long long)blockIdx.x * blockDim.x + threadIdx.x;
    if (i < n) p[i] = 0.0f;
}

// ---------------------------------------------------------------------------
// BatchNorm: column stats (one block per column) + apply
// ---------------------------------------------------------------------------
__global__ __launch_bounds__(256)
void k_colstats(const float* __restrict__ X, int rows, int ld,
                float* __restrict__ stats, int cols) {
    const int c = blockIdx.x;
    __shared__ float ssum[256], ssq[256];
    float s = 0.0f, q = 0.0f;
    for (int r = threadIdx.x; r < rows; r += 256) {
        float v = X[(long long)r * ld + c];
        s += v; q += v * v;
    }
    ssum[threadIdx.x] = s; ssq[threadIdx.x] = q;
    __syncthreads();
    for (int off = 128; off > 0; off >>= 1) {
        if (threadIdx.x < off) {
            ssum[threadIdx.x] += ssum[threadIdx.x + off];
            ssq[threadIdx.x]  += ssq[threadIdx.x + off];
        }
        __syncthreads();
    }
    if (threadIdx.x == 0) { stats[c] = ssum[0]; stats[cols + c] = ssq[0]; }
}

__global__ void k_bn_apply(const float* __restrict__ X, int ldin,
                           const float* __restrict__ stats, int cols, float invRows,
                           const float* __restrict__ g, const float* __restrict__ b,
                           float* __restrict__ Y, int ldout, int relu) {
    int n = blockIdx.x, c = threadIdx.x;
    if (c >= cols) return;
    float mean = stats[c] * invRows;
    float var  = stats[cols + c] * invRows - mean * mean;
    float inv  = rsqrtf(var + 1e-5f);
    float v = g[c] * (X[(long long)n * ldin + c] - mean) * inv + b[c];
    if (relu) v = fmaxf(v, 0.0f);
    Y[(long long)n * ldout + c] = v;
}

// ---------------------------------------------------------------------------
// WMMA GEMM:  C[M,NP] = maybe_relu(A[M,K](f16) * B[K,NP](f16) + bias)
// K % 32 == 0, NP % 64 == 0.  Block 256 thr = 8 waves, tile 256(M) x 64(N);
// wave w owns rows [w*32, w*32+32): 2 A-frags x 4 B-frags -> 8 v_wmma per
// wave per K-step.  All fragment ds loads hoisted ahead of the WMMA burst;
// double-buffered LDS -> one barrier per K-step; global_prefetch_b8 ahead.
// ---------------------------------------------------------------------------
template<bool OUT_F16, bool RELU>
__global__ __launch_bounds__(256)
void k_gemm_wmma(const _Float16* __restrict__ A, const _Float16* __restrict__ Bw,
                 const float* __restrict__ bias, int nbias,
                 void* __restrict__ Cout, int M, int K, int NP) {
    __shared__ _Float16 As[2][256 * 32];   // [m][k]
    __shared__ _Float16 Bt[2][64 * 32];    // [n][k] (transposed)
    const int tid  = threadIdx.x;
    const int lane = tid & 31;
    const int wave = tid >> 5;             // owns rows [wave*32, wave*32+32)
    const int m0   = blockIdx.x * 256;
    const int n0   = blockIdx.y * 64;
    const int nl   = lane & 15;
    const int hs   = lane >> 4;

    // 2(M) x 4(N) accumulators, bias folded in
    v8f acc[2][4];
    #pragma unroll
    for (int j = 0; j < 4; ++j) {
        int col = n0 + j * 16 + nl;
        float bv = (bias && col < nbias) ? bias[col] : 0.0f;
        #pragma unroll
        for (int t = 0; t < 2; ++t)
            #pragma unroll
            for (int i = 0; i < 8; ++i) acc[t][j][i] = bv;
    }

    // staging: A one row/thread (32 halves = 4 x v8h), B 8 halves/thread
    const int ar = tid;                    // 0..255
    const int bk = tid >> 3;               // 0..31
    const int bc = (tid & 7) * 8;          // 0..56
    const bool arow_ok = (m0 + ar) < M;
    const long long abase_g = (long long)(m0 + ar) * K;

    // ---- preload tile 0 ----
    {
        v8h a[4] = {};
        if (arow_ok) {
            #pragma unroll
            for (int c = 0; c < 4; ++c) a[c] = *(const v8h*)(A + abase_g + c * 8);
        }
        #pragma unroll
        for (int c = 0; c < 4; ++c) *(v8h*)(&As[0][ar * 32 + c * 8]) = a[c];
        v8h bv = *(const v8h*)(Bw + (long long)bk * NP + n0 + bc);
        #pragma unroll
        for (int j = 0; j < 8; ++j) Bt[0][(bc + j) * 32 + bk] = bv[j];
    }
    __syncthreads();

    int buf = 0;
    for (int k0 = 0; k0 < K; k0 += 32) {
        const int  nxt  = buf ^ 1;
        const bool more = (k0 + 32) < K;

        // issue global loads for next tile while computing current one
        v8h na[4] = {}, nb = {};
        if (more) {
            if (arow_ok) {
                #pragma unroll
                for (int c = 0; c < 4; ++c)
                    na[c] = *(const v8h*)(A + abase_g + k0 + 32 + c * 8);
            }
            nb = *(const v8h*)(Bw + (long long)(k0 + 32 + bk) * NP + n0 + bc);
            if (k0 + 64 < K) {   // global_prefetch_b8 two tiles ahead
                __builtin_prefetch(A + abase_g + k0 + 64, 0, 1);
                __builtin_prefetch(Bw + (long long)(k0 + 64 + bk) * NP + n0 + bc, 0, 1);
            }
        }

        // hoist ALL fragment loads (CDNA5 16-bit layout: 2 contiguous 16B runs)
        v16h af[2];
        #pragma unroll
        for (int t = 0; t < 2; ++t) {
            const int afb = ((wave * 2 + t) * 16 + nl) * 32 + hs * 8;
            v8h fa0 = *(const v8h*)(&As[buf][afb]);
            v8h fa1 = *(const v8h*)(&As[buf][afb + 16]);
            af[t] = __builtin_shufflevector(fa0, fa1, 0,1,2,3,4,5,6,7,8,9,10,11,12,13,14,15);
        }
        v16h bf[4];
        #pragma unroll
        for (int j = 0; j < 4; ++j) {
            const int bfb = (j * 16 + nl) * 32 + hs * 8;
            v8h fb0 = *(const v8h*)(&Bt[buf][bfb]);
            v8h fb1 = *(const v8h*)(&Bt[buf][bfb + 16]);
            bf[j] = __builtin_shufflevector(fb0, fb1, 0,1,2,3,4,5,6,7,8,9,10,11,12,13,14,15);
        }

        // WMMA burst: 8 independent accumulators, back-to-back issue
        #pragma unroll
        for (int t = 0; t < 2; ++t)
            #pragma unroll
            for (int j = 0; j < 4; ++j)
                acc[t][j] = __builtin_amdgcn_wmma_f32_16x16x32_f16(
                    false, af[t], false, bf[j], (short)0, acc[t][j], false, false);

        // stash next tile into the other buffer, then one barrier
        if (more) {
            #pragma unroll
            for (int c = 0; c < 4; ++c) *(v8h*)(&As[nxt][ar * 32 + c * 8]) = na[c];
            #pragma unroll
            for (int j = 0; j < 8; ++j) Bt[nxt][(bc + j) * 32 + bk] = nb[j];
        }
        __syncthreads();
        buf = nxt;
    }

    // store: C layout -> row = r + hs*8 within each 16-row tile
    #pragma unroll
    for (int t = 0; t < 2; ++t) {
        #pragma unroll
        for (int j = 0; j < 4; ++j) {
            int col = n0 + j * 16 + nl;
            #pragma unroll
            for (int r = 0; r < 8; ++r) {
                int row = m0 + (wave * 2 + t) * 16 + hs * 8 + r;
                if (row < M) {
                    float v = acc[t][j][r];
                    if (RELU) v = fmaxf(v, 0.0f);
                    if (OUT_F16) ((_Float16*)Cout)[(long long)row * NP + col] = (_Float16)v;
                    else         ((float*)Cout)[(long long)row * NP + col]    = v;
                }
            }
        }
    }
}

// ---------------------------------------------------------------------------
extern "C" void kernel_launch(void* const* d_in, const int* in_sizes, int n_in,
                              void* d_out, int out_size, void* d_ws, size_t ws_size,
                              hipStream_t stream) {
    const int N  = in_sizes[0];
    const int E  = in_sizes[2];
    const int Bg = out_size / H;          // batch_size (device scalar unreadable in capture)

    const int*   x        = (const int*)d_in[0];
    const int*   ei       = (const int*)d_in[1];
    const int*   src      = ei;
    const int*   dst      = ei + E;
    const int*   eattr    = (const int*)d_in[2];
    const int*   batch    = (const int*)d_in[3];
    const float* atom_emb = (const float*)d_in[5];
    const float* bond_emb = (const float*)d_in[6];
    const float* vn_emb   = (const float*)d_in[7];
    const float* W1   = (const float*)d_in[8];
    const float* b1   = (const float*)d_in[9];
    const float* W2   = (const float*)d_in[10];
    const float* b2   = (const float*)d_in[11];
    const float* bn_g = (const float*)d_in[12];
    const float* bn_b = (const float*)d_in[13];
    const float* vW1  = (const float*)d_in[14];
    const float* vb1  = (const float*)d_in[15];
    const float* vg1  = (const float*)d_in[16];
    const float* vbb1 = (const float*)d_in[17];
    const float* vW2  = (const float*)d_in[18];
    const float* vb2  = (const float*)d_in[19];
    const float* vg2  = (const float*)d_in[20];
    const float* vbb2 = (const float*)d_in[21];
    (void)n_in; (void)ws_size;

    // -------- workspace partition --------
    char* p = (char*)d_ws;
    auto alloc = [&](size_t bytes) { void* r = (void*)p; p += (bytes + 255) & ~(size_t)255; return r; };
    float*    ef      = (float*)   alloc((size_t)E  * H   * 4);
    float*    h       = (float*)   alloc((size_t)N  * H   * 4);
    float*    hin     = (float*)   alloc((size_t)N  * H   * 4);
    float*    aggr    = (float*)   alloc((size_t)N  * H   * 4);
    float*    outf    = (float*)   alloc((size_t)N  * HP  * 4);
    _Float16* preh    = (_Float16*)alloc((size_t)N  * HP  * 2);
    _Float16* th      = (_Float16*)alloc((size_t)N  * H2P * 2);
    _Float16* W1h     = (_Float16*)alloc((size_t)NLAYER * HP  * H2P * 2);
    _Float16* W2h     = (_Float16*)alloc((size_t)NLAYER * H2P * HP  * 2);
    _Float16* vW1h    = (_Float16*)alloc((size_t)HP  * H2P * 2);
    _Float16* vW2h    = (_Float16*)alloc((size_t)H2P * HP  * 2);
    float*    vn      = (float*)   alloc((size_t)Bg * H   * 4);
    float*    pooled  = (float*)   alloc((size_t)Bg * H   * 4);
    _Float16* pooledh = (_Float16*)alloc((size_t)Bg * HP  * 2);
    float*    vt1     = (float*)   alloc((size_t)Bg * H2P * 4);
    _Float16* vth     = (_Float16*)alloc((size_t)Bg * H2P * 2);
    float*    vt2     = (float*)   alloc((size_t)Bg * HP  * 4);
    float*    stats   = (float*)   alloc((size_t)2 * H2P * 4);

    // -------- weights -> padded f16 --------
    for (int i = 0; i < NLAYER; ++i) {
        k_castpad<<<HP,  H2P, 0, stream>>>(W1 + (size_t)i*H*H2,  H,  H2, H2, W1h + (size_t)i*HP*H2P, H2P);
        k_castpad<<<H2P, HP,  0, stream>>>(W2 + (size_t)i*H2*H,  H2, H,  H,  W2h + (size_t)i*H2P*HP, HP);
    }
    k_castpad<<<HP,  H2P, 0, stream>>>(vW1, H,  H2, H2, vW1h, H2P);
    k_castpad<<<H2P, HP,  0, stream>>>(vW2, H2, H,  H,  vW2h, HP);

    // -------- encoders + virtual-node init --------
    k_gather<<<N,  HP, 0, stream>>>(x,     atom_emb, h);
    k_gather<<<E,  HP, 0, stream>>>(eattr, bond_emb, ef);
    k_init_vn<<<Bg, HP, 0, stream>>>(vn_emb, vn);

    const int mg  = (N  + 255) / 256;
    const int bgt = (Bg + 255) / 256;

    for (int i = 0; i < NLAYER; ++i) {
        // GINE message passing
        k_addvn  <<<N, HP, 0, stream>>>(h, vn, batch, hin, aggr);
        k_message<<<E, HP, 0, stream>>>(src, dst, hin, ef, aggr);
        k_pre_cast<<<N, HP, 0, stream>>>(hin, aggr, preh);

        // node MLP: relu(pre@W1+b1) @ W2 + b2   (WMMA f16->f32)
        k_gemm_wmma<true,  true ><<<dim3(mg, H2P/64), 256, 0, stream>>>(
            preh, W1h + (size_t)i*HP*H2P, b1 + (size_t)i*H2, H2, th,   N, HP,  H2P);
        k_gemm_wmma<false, false><<<dim3(mg, HP/64),  256, 0, stream>>>(
            th,   W2h + (size_t)i*H2P*HP, b2 + (size_t)i*H,  H,  outf, N, H2P, HP);

        // BatchNorm (+relu except last layer) -> h
        k_colstats<<<H, 256, 0, stream>>>(outf, N, HP, stats, H);
        k_bn_apply<<<N, HP, 0, stream>>>(outf, HP, stats, H, 1.0f/(float)N,
            bn_g + (size_t)i*H, bn_b + (size_t)i*H, h, H, (i < NLAYER-1) ? 1 : 0);

        if (i < NLAYER - 1) {
            // virtual-node update
            long long pn = (long long)Bg * H;
            k_zero<<<(int)((pn + 255)/256), 256, 0, stream>>>(pooled, pn);
            k_pool<<<N, HP, 0, stream>>>(batch, h, pooled);
            k_pooled_cast<<<Bg, HP, 0, stream>>>(pooled, vn, pooledh);

            k_gemm_wmma<false, false><<<dim3(bgt, H2P/64), 256, 0, stream>>>(
                pooledh, vW1h, vb1, H2, vt1, Bg, HP, H2P);
            k_colstats<<<H2, 256, 0, stream>>>(vt1, Bg, H2P, stats, H2);
            k_bn_apply<<<Bg, H2P, 0, stream>>>(vt1, H2P, stats, H2, 1.0f/(float)Bg,
                                               vg1, vbb1, vt1, H2P, 1);
            k_castpad<<<Bg, H2P, 0, stream>>>(vt1, Bg, H2P, H2P, vth, H2P);

            k_gemm_wmma<false, false><<<dim3(bgt, HP/64), 256, 0, stream>>>(
                vth, vW2h, vb2, H, vt2, Bg, H2P, HP);
            k_colstats<<<H, 256, 0, stream>>>(vt2, Bg, HP, stats, H);
            k_bn_apply<<<Bg, HP, 0, stream>>>(vt2, HP, stats, H, 1.0f/(float)Bg,
                                              vg2, vbb2, vn, H, 1);
        }
    }

    // readout: global add pool -> d_out [B,H]
    long long on = (long long)Bg * H;
    k_zero<<<(int)((on + 255)/256), 256, 0, stream>>>((float*)d_out, on);
    k_pool<<<N, HP, 0, stream>>>(batch, h, (float*)d_out);
}
